// QuantumFrontEndQCNN_12524124635735
// MI455X (gfx1250) — compile-verified, matrix operand order checked by reading, not run on hardware
//
#include <hip/hip_runtime.h>
#include <math.h>

// ---------------- problem constants ----------------
#define NB    4          // batch
#define CCH   64         // channels
#define HH    128
#define WW    128
#define LPB   4096       // Ho*Wo patches per image
#define RTOT  16384      // NB * LPB rows
#define PD    576        // C*9 patch features
#define NQD   6
#define QDIM  256        // 2^8 statevector
#define COUT  64
#define KTOT  832        // QDIM + PD, packed weight K extent
#define PI_F  3.14159265358979f

typedef __attribute__((ext_vector_type(16))) __bf16 v16bf;
typedef __attribute__((ext_vector_type(8)))  __bf16 v8bf;
typedef __attribute__((ext_vector_type(8)))  float  v8f;
typedef __attribute__((ext_vector_type(4)))  unsigned int u32x4;
typedef __attribute__((ext_vector_type(8)))  int  i32x8;
typedef __attribute__((ext_vector_type(4)))  int  i32x4;

__device__ __forceinline__ float shflx(float v, int m) { return __shfl_xor(v, m, 32); }

// ---------------- kernel 0: prep (packed bf16 weights, style angles, fused bias) ---
// wcat[n][0:256] = out_proj_w[n], wcat[n][256:832] = res_proj_w[n]
__global__ __launch_bounds__(256)
void prep_kernel(const float* __restrict__ style,      // (4,128)
                 const float* __restrict__ sw,         // (6,128)
                 const float* __restrict__ sb,         // (6)
                 const float* __restrict__ opw,        // (64,256)
                 const float* __restrict__ opb,        // (64)
                 const float* __restrict__ rpw,        // (64,576)
                 const float* __restrict__ rpb,        // (64)
                 __bf16* __restrict__ wcat,            // (64,832)
                 float* __restrict__ sa,               // (4,6)
                 float* __restrict__ bias_c)           // (64)
{
    const int t = threadIdx.x;
    for (int i = t; i < COUT * KTOT; i += 256) {
        const int n = i / KTOT, k = i - n * KTOT;
        const float w = (k < QDIM) ? opw[n * QDIM + k] : rpw[n * PD + (k - QDIM)];
        wcat[i] = (__bf16)w;
    }
    if (t < COUT) bias_c[t] = opb[t] + rpb[t];
    if (t < NB * NQD) {
        const int b = t / NQD, i = t % NQD;
        float acc = sb[i];
        for (int k = 0; k < 128; ++k) acc += style[b * 128 + k] * sw[i * 128 + k];
        sa[t] = PI_F * tanhf(acc);
    }
}

// ---------------- kernel 1: unfold -> pf(bf16) + init angles --------------------
__global__ __launch_bounds__(256)
void pf_angles_kernel(const float* __restrict__ x,          // (4,64,128,128)
                      const float* __restrict__ inp_scale,  // (576)
                      const float* __restrict__ dpw,        // (6,576)
                      const float* __restrict__ dpb,        // (6)
                      const float* __restrict__ sa,         // (4,6)
                      __bf16* __restrict__ pf,              // (16384,576)
                      float* __restrict__ ang)              // (16384,6)
{
    const int lane = threadIdx.x & 31;
    const int row  = blockIdx.x * 8 + (threadIdx.x >> 5);
    const int b    = row >> 12;
    const int p    = row & (LPB - 1);
    const int oy   = p >> 6;
    const int ox   = p & 63;

    float acc[NQD];
#pragma unroll
    for (int i = 0; i < NQD; ++i) acc[i] = 0.0f;

#pragma unroll 2
    for (int t = 0; t < PD / 32; ++t) {
        const int col = t * 32 + lane;
        const int c  = col / 9;
        const int k  = col - c * 9;
        const int ki = k / 3;
        const int kj = k - ki * 3;
        const int y  = oy * 2 - 1 + ki;
        const int xx = ox * 2 - 1 + kj;
        float v = 0.0f;
        if (y >= 0 && y < HH && xx >= 0 && xx < WW)
            v = x[(((size_t)b * CCH + c) * HH + y) * WW + xx];
        v *= inp_scale[col];
        pf[(size_t)row * PD + col] = (__bf16)v;
#pragma unroll
        for (int i = 0; i < NQD; ++i) acc[i] += v * dpw[i * PD + col];
    }
#pragma unroll
    for (int i = 0; i < NQD; ++i) {
        float a = acc[i];
        a += shflx(a, 16); a += shflx(a, 8); a += shflx(a, 4);
        a += shflx(a, 2);  a += shflx(a, 1);
        acc[i] = a;
    }
    if (lane == 0) {
#pragma unroll
        for (int i = 0; i < NQD; ++i)
            ang[row * NQD + i] = PI_F * tanhf(acc[i] + dpb[i]) + sa[b * NQD + i];
    }
}

// ---------------- kernel 2: 8-qubit statevector per row, wave32-resident ---------
__device__ __forceinline__ void ry_xlane(float (&re)[8], float (&im)[8],
                                         int lane, int w, float c, float s) {
    const int mask = 1 << (4 - w);
    const float ss = (lane & mask) ? s : -s;
#pragma unroll
    for (int j = 0; j < 8; ++j) {
        float pr = shflx(re[j], mask), pi = shflx(im[j], mask);
        re[j] = c * re[j] + ss * pr;
        im[j] = c * im[j] + ss * pi;
    }
}

__device__ __forceinline__ void ry_w5(float (&re)[8], float (&im)[8], float c, float s) {
#pragma unroll
    for (int j = 0; j < 4; ++j) {
        float a0r = re[j], a1r = re[j + 4];
        float a0i = im[j], a1i = im[j + 4];
        re[j]     = c * a0r - s * a1r;  im[j]     = c * a0i - s * a1i;
        re[j + 4] = s * a0r + c * a1r;  im[j + 4] = s * a0i + c * a1i;
    }
}

__device__ __forceinline__ void ry_any(float (&re)[8], float (&im)[8],
                                       int lane, int w, float theta) {
    const float c = __cosf(0.5f * theta), s = __sinf(0.5f * theta);
    if (w <= 4) ry_xlane(re, im, lane, w, c, s);
    else        ry_w5(re, im, c, s);
}

__device__ __forceinline__ void rz_any(float (&re)[8], float (&im)[8],
                                       int lane, int w, float theta) {
    const float c = __cosf(0.5f * theta), s = __sinf(0.5f * theta);
#pragma unroll
    for (int j = 0; j < 8; ++j) {
        const int bit = (w <= 4) ? ((lane >> (4 - w)) & 1) : ((j >> 2) & 1);
        const float sg = bit ? -s : s;
        float r = re[j], i = im[j];
        re[j] = c * r + sg * i;
        im[j] = c * i - sg * r;
    }
}

__device__ __forceinline__ void cnot_xlane(float (&re)[8], float (&im)[8],
                                           int lane, int ctrl, int tgt) {
    const int tmask = 1 << (4 - tgt);
#pragma unroll
    for (int j = 0; j < 8; ++j) {
        float pr = shflx(re[j], tmask), pi = shflx(im[j], tmask);
        const int cb = (ctrl <= 4) ? ((lane >> (4 - ctrl)) & 1) : ((j >> 2) & 1);
        if (cb) { re[j] = pr; im[j] = pi; }
    }
}

__device__ __forceinline__ void cnot_45(float (&re)[8], float (&im)[8], int lane) {
    if (lane & 1) {
#pragma unroll
        for (int j = 0; j < 4; ++j) {
            float tr = re[j]; re[j] = re[j + 4]; re[j + 4] = tr;
            float ti = im[j]; im[j] = im[j + 4]; im[j + 4] = ti;
        }
    }
}

__global__ __launch_bounds__(256)
void circuit_kernel(const float* __restrict__ ang,    // (16384,6)
                    const float* __restrict__ qcnn,   // (2,6,2,3)
                    const float* __restrict__ mp,     // (6,3)
                    __bf16* __restrict__ probs)       // (16384,256)
{
    const int lane = threadIdx.x & 31;
    const int row  = blockIdx.x * 8 + (threadIdx.x >> 5);

    float re[8], im[8];
#pragma unroll
    for (int j = 0; j < 8; ++j) { re[j] = 0.0f; im[j] = 0.0f; }
    if (lane == 0) re[0] = 1.0f;

#pragma unroll
    for (int i = 0; i < NQD; ++i) ry_any(re, im, lane, i, ang[row * NQD + i]);

#pragma unroll
    for (int l = 0; l < 2; ++l) {
#pragma unroll
        for (int i = 0; i < NQD; ++i) {
            ry_any(re, im, lane, i, qcnn[((l * NQD + i) * 2 + 0) * 3]);
            rz_any(re, im, lane, i, qcnn[((l * NQD + i) * 2 + 1) * 3]);
        }
        cnot_xlane(re, im, lane, 0, 1);
        cnot_xlane(re, im, lane, 1, 2);
        cnot_xlane(re, im, lane, 2, 3);
        cnot_xlane(re, im, lane, 3, 4);
        cnot_45(re, im, lane);
        cnot_xlane(re, im, lane, 5, 0);
    }

#pragma unroll
    for (int i = 0; i < NQD; ++i) {
        rz_any(re, im, lane, i, mp[i * 3 + 0]);
        ry_any(re, im, lane, i, mp[i * 3 + 1]);
        rz_any(re, im, lane, i, mp[i * 3 + 2]);
    }

    v8bf p;
#pragma unroll
    for (int j = 0; j < 8; ++j) p[j] = (__bf16)(re[j] * re[j] + im[j] * im[j]);
    *(v8bf*)(probs + (size_t)row * QDIM + lane * 8) = p;
}

// ---------------- kernel 3: fused dual-GEMM, TDM-staged B tile + WMMA bf16 -------
// Block owns N-tile nt (16 cols) and 8 M-tiles (one per wave). TDM stages the
// 16x832 bf16 B tile (packed wq|wr rows) into LDS once; WMMA B fragments come
// from ds_load_b128, A fragments stream from global.
__global__ __launch_bounds__(256)
void gemm_kernel(const __bf16* __restrict__ probs,  // (16384,256)
                 const __bf16* __restrict__ pf,     // (16384,576)
                 const __bf16* __restrict__ wcat,   // (64,832) packed weights
                 const float* __restrict__ bias_c,  // (64)
                 float* __restrict__ out)           // (4,64,64,64)
{
    __shared__ __align__(16) __bf16 lb[16][KTOT];   // 26,624 bytes

    const int lane = threadIdx.x & 31;
    const int wave = threadIdx.x >> 5;
    const int nt   = blockIdx.x & 3;
    const int mt   = (blockIdx.x >> 2) * 8 + wave;
    const int m0   = mt * 16;
    const int n0   = nt * 16;

    // ---- TDM: stage B tile rows n0..n0+15 of wcat into LDS (wave 0 issues) ----
    if (threadIdx.x < 32) {
        const unsigned long long ga =
            (unsigned long long)(uintptr_t)(wcat + (size_t)n0 * KTOT);
        const unsigned int lva = (unsigned int)(uintptr_t)(&lb[0][0]);
        u32x4 g0;
        g0[0] = 1u;                                       // count=1, user mode
        g0[1] = lva;                                      // lds_addr
        g0[2] = (unsigned int)(ga & 0xFFFFFFFFu);         // global_addr[31:0]
        g0[3] = (unsigned int)((ga >> 32) & 0x01FFFFFFu)  // global_addr[56:32]
                | (2u << 30);                             // type=2 (image)
        i32x8 g1;
        g1[0] = (int)(1u << 16);              // data_size=1 -> 2-byte elements
        g1[1] = (int)((unsigned)KTOT << 16);  // tensor_dim0[15:0]
        g1[2] = (int)(16u << 16);             // tensor_dim0[31:16]=0 | tensor_dim1=16
        g1[3] = (int)((unsigned)KTOT << 16);  // tensor_dim1[31:16]=0 | tile_dim0
        g1[4] = 16;                           // tile_dim1=16, tile_dim2=0
        g1[5] = KTOT;                         // tensor_dim0_stride[31:0]
        g1[6] = 0;                            // stride hi | tensor_dim1_stride lo
        g1[7] = 0;
        i32x4 z4 = {0, 0, 0, 0};
#if defined(__clang_major__) && (__clang_major__ >= 23)
        i32x8 z8 = {0, 0, 0, 0, 0, 0, 0, 0};
        __builtin_amdgcn_tensor_load_to_lds(g0, g1, z4, z4, z8, 0);
#else
        __builtin_amdgcn_tensor_load_to_lds(g0, g1, z4, z4, 0);
#endif
        __builtin_amdgcn_s_wait_tensorcnt(0);
    }
    __syncthreads();
    asm volatile("" ::: "memory");   // LDS now holds the B tile

    const int mrow = m0 + (lane & 15);
    const int lrow = lane & 15;                 // local B row in LDS
    const int kb   = (lane >> 4) * 8;           // 16-bit fragment layout

    union Frag { v16bf v; v8bf h[2]; };
    v8f acc = {};

    // GEMM 1: probs @ wq^T  (B rows = lb[.][0:256])
#pragma unroll
    for (int k0 = 0; k0 < QDIM; k0 += 32) {
        Frag a, bm;
        a.h[0]  = *(const v8bf*)(probs + (size_t)mrow * QDIM + k0 + kb);
        a.h[1]  = *(const v8bf*)(probs + (size_t)mrow * QDIM + k0 + 16 + kb);
        bm.h[0] = *(const v8bf*)(&lb[lrow][k0 + kb]);
        bm.h[1] = *(const v8bf*)(&lb[lrow][k0 + 16 + kb]);
        acc = __builtin_amdgcn_wmma_f32_16x16x32_bf16(
                  false, a.v, false, bm.v, (short)0, acc, false, false);
    }

    // GEMM 2: pf @ wr^T  (B rows = lb[.][256:832])
#pragma unroll
    for (int k0 = 0; k0 < PD; k0 += 32) {
        if (k0 + 128 < PD)
            __builtin_prefetch(pf + (size_t)mrow * PD + k0 + 128, 0, 1);
        Frag a, bm;
        a.h[0]  = *(const v8bf*)(pf + (size_t)mrow * PD + k0 + kb);
        a.h[1]  = *(const v8bf*)(pf + (size_t)mrow * PD + k0 + 16 + kb);
        bm.h[0] = *(const v8bf*)(&lb[lrow][QDIM + k0 + kb]);
        bm.h[1] = *(const v8bf*)(&lb[lrow][QDIM + k0 + 16 + kb]);
        acc = __builtin_amdgcn_wmma_f32_16x16x32_bf16(
                  false, a.v, false, bm.v, (short)0, acc, false, false);
    }

    // C/D layout: lane holds n = lane&15, rows m = m0 + (lane>>4)*8 + r
    const int n  = n0 + (lane & 15);
    const float bn = bias_c[n];
    const int mb = m0 + (lane >> 4) * 8;
    const int b  = mb >> 12;
    const int ml = mb & (LPB - 1);
    float* o = out + ((size_t)(b * COUT + n)) * LPB + ml;
    float4 r0 = make_float4(acc[0] + bn, acc[1] + bn, acc[2] + bn, acc[3] + bn);
    float4 r1 = make_float4(acc[4] + bn, acc[5] + bn, acc[6] + bn, acc[7] + bn);
    *(float4*)(o)     = r0;
    *(float4*)(o + 4) = r1;
}

// ---------------- launch ---------------------------------------------------------
extern "C" void kernel_launch(void* const* d_in, const int* in_sizes, int n_in,
                              void* d_out, int out_size, void* d_ws, size_t ws_size,
                              hipStream_t stream) {
    const float* x    = (const float*)d_in[0];
    const float* sty  = (const float*)d_in[1];
    const float* insc = (const float*)d_in[2];
    const float* dpw  = (const float*)d_in[3];
    const float* dpb  = (const float*)d_in[4];
    const float* sw   = (const float*)d_in[5];
    const float* sb   = (const float*)d_in[6];
    const float* qcnn = (const float*)d_in[7];
    const float* mp   = (const float*)d_in[8];
    const float* opw  = (const float*)d_in[9];
    const float* opb  = (const float*)d_in[10];
    const float* rpw  = (const float*)d_in[11];
    const float* rpb  = (const float*)d_in[12];
    float* out = (float*)d_out;

    char* ws = (char*)d_ws;
    size_t off = 0;
    __bf16* pf_bf    = (__bf16*)(ws + off); off += (size_t)RTOT * PD * 2;     // 18.9 MB
    __bf16* probs_bf = (__bf16*)(ws + off); off += (size_t)RTOT * QDIM * 2;   // 8.4 MB
    float*  ang      = (float*) (ws + off); off += (size_t)RTOT * NQD * 4;    // 0.4 MB
    __bf16* wcat     = (__bf16*)(ws + off); off += (size_t)COUT * KTOT * 2;   // 104 KB
    float*  sa       = (float*) (ws + off); off += 256;
    float*  bias_c   = (float*) (ws + off); off += 256;

    prep_kernel<<<1, 256, 0, stream>>>(sty, sw, sb, opw, opb, rpw, rpb,
                                       wcat, sa, bias_c);
    pf_angles_kernel<<<RTOT / 8, 256, 0, stream>>>(x, insc, dpw, dpb, sa, pf_bf, ang);
    circuit_kernel<<<RTOT / 8, 256, 0, stream>>>(ang, qcnn, mp, probs_bf);
    // 512 blocks: 4 N-tiles x 128 groups of 8 M-tiles
    gemm_kernel<<<(RTOT / 16 / 8) * 4, 256, 0, stream>>>(probs_bf, pf_bf, wcat,
                                                         bias_c, out);
}
